// DCSAB_51513837748612
// MI455X (gfx1250) — matrix operand order repaired and checked by reading, not compiled, not run on hardware
//
#include <hip/hip_runtime.h>
#include <hip/hip_bf16.h>
#include <stddef.h>

// ---------------------------------------------------------------------------
// DCSAB fused pipeline for MI455X (gfx1250, wave32, WMMA)
//   x:[8,256,224,224] f32 -> avgpool7 -> GN(1,C) -> depthwise QKV (h & w)
//   -> channel attention -> gate = sigmoid(mean attn) -> out = gate * x
// Roofline: ~1.23 GB HBM traffic -> ~53us floor. Middle stage fused into one
// WMMA workgroup per (batch, head); mean_n(P.V) = P.mean_n(V) collapses the
// second GEMM to a mat-vec, leaving QK^T as the WMMA core.
// All convs run guard-free on zero-halo LDS (w-direction via transposition),
// so WMMA waves keep EXEC full with no divergent branches.
// ---------------------------------------------------------------------------

typedef __attribute__((ext_vector_type(16))) _Float16 v16h;
typedef __attribute__((ext_vector_type(8)))  float    v8f;

#define B_    8
#define C_    256
#define H_    224
#define W_    224
#define PH    32      // pooled H (= pooled W)
#define PW    32
#define HEADS 8
#define DHEAD 32      // C/HEADS
#define HALO  10      // max conv radius (v: 21 taps)
#define YROWS (PH + 2*HALO)      // 52 rows incl. zero halo
#define YSTR  33                 // padded row stride (odd -> conflict-free)
#define YCH   (YROWS*YSTR)       // 1716 floats per channel in LDS

// ---------------- Kernel 1: 7x7 avg pool + per-block GN partial sums -------
// grid = B*C*PH blocks, 256 threads. Coalesced 224-float row reads.
__global__ __launch_bounds__(256)
void k_pool_stats(const float* __restrict__ x, float* __restrict__ y,
                  float* __restrict__ partial) {
    const int blk = blockIdx.x;
    const int oh  = blk & 31;
    const int bc  = blk >> 5;            // b*256 + c
    const int b   = bc >> 8;
    const float* xrow = x + ((size_t)bc * H_ + (size_t)oh * 7) * W_;
    __shared__ float colsum[W_];
    const int t = threadIdx.x;
    if (t < W_) {
        float s = 0.f;
        #pragma unroll
        for (int r = 0; r < 7; ++r) s += xrow[(size_t)r * W_ + t];
        colsum[t] = s;
    }
    __syncthreads();
    if (t < PW) {
        float s = 0.f;
        #pragma unroll
        for (int u = 0; u < 7; ++u) s += colsum[t * 7 + u];
        const float p = s * (1.f / 49.f);
        y[(size_t)bc * (PH * PW) + oh * PW + t] = p;
        float s1 = p, s2 = p * p;                 // wave32 tree reduce
        for (int off = 16; off; off >>= 1) {
            s1 += __shfl_down(s1, off);
            s2 += __shfl_down(s2, off);
        }
        if (t == 0) {
            const int pidx = b * 8192 + (bc & 255) * 32 + oh;
            partial[2 * pidx]     = s1;
            partial[2 * pidx + 1] = s2;
        }
    }
}

// ---------------- Kernel 2: reduce partials -> per-batch mean, rstd --------
__global__ __launch_bounds__(256)
void k_gn_stats(const float* __restrict__ partial, float* __restrict__ stats) {
    const int b = blockIdx.x, t = threadIdx.x;
    const float* p = partial + (size_t)b * 8192 * 2;
    float s1 = 0.f, s2 = 0.f;
    for (int i = t; i < 8192; i += 256) { s1 += p[2 * i]; s2 += p[2 * i + 1]; }
    __shared__ float r1[256], r2[256];
    r1[t] = s1; r2[t] = s2; __syncthreads();
    for (int off = 128; off; off >>= 1) {
        if (t < off) { r1[t] += r1[t + off]; r2[t] += r2[t + off]; }
        __syncthreads();
    }
    if (t == 0) {
        const float inv = 1.f / (float)(C_ * PH * PW);
        const float mean = r1[0] * inv;
        const float var  = r2[0] * inv - mean * mean;
        stats[2 * b]     = mean;
        stats[2 * b + 1] = rsqrtf(var + 1e-5f);
    }
}

// ------------- guard-free 1-D conv down rows of a zero-halo channel --------
// logical row i in [0,31] lives at LDS row (i+HALO); halo rows are zero, so
// any radius R<=HALO needs no bounds checks: pure ds_load + v_fmac.
template<int R>
__device__ __forceinline__ float dconvH(const float* __restrict__ ych,
                                        int i, int j, const float* w) {
    const float* p = ych + (i + HALO - R) * YSTR + j;
    float s = 0.f;
    #pragma unroll
    for (int t = 0; t < 2 * R + 1; ++t) s = fmaf(w[t], p[t * YSTR], s);
    return s;
}

// ---------------- Kernel 3: per-(b,head) WMMA attention -> gate ------------
// 64 blocks x 256 threads (8 waves). Dynamic LDS (~224.5 KB):
//   ys[32*1716] zero-halo y (dir 0: as-is, dir 1: transposed)
//   scores[32*33] | vpart[128] | vbar[32]
__global__ __launch_bounds__(256)
void k_attn_gate(const float* __restrict__ yg, const float* __restrict__ stats,
                 const float* __restrict__ gamma, const float* __restrict__ beta,
                 const float* __restrict__ wqh, const float* __restrict__ wkh,
                 const float* __restrict__ wvh, const float* __restrict__ wqw,
                 const float* __restrict__ wkw, const float* __restrict__ wvw,
                 float* __restrict__ gate) {
    extern __shared__ float smem[];
    float* ys     = smem;                    // 32*1716 = 54912
    float* scores = ys + DHEAD * YCH;        // 32*33   = 1056
    float* vpart  = scores + DHEAD * YSTR;   // 128
    float* vbar   = vpart + 128;             // 32

    const int b    = blockIdx.x >> 3;
    const int head = blockIdx.x & 7;
    const int c0   = head * DHEAD;
    const int tid  = threadIdx.x;
    const int wid  = tid >> 5;
    const int lane = tid & 31;

    const float mu = stats[2 * b], rstd = stats[2 * b + 1];
    float gacc = 0.f;                         // meaningful for tid < 32

    for (int dir = 0; dir < 2; ++dir) {       // 0 = h-direction, 1 = w (transposed)
        // --- zero the halo'd channel buffers ---
        for (int idx = tid; idx < DHEAD * YCH; idx += 256) ys[idx] = 0.f;
        __syncthreads();
        // --- load + normalize y slice (L2-resident); dir 1 stores transposed ---
        for (int idx = tid; idx < DHEAD * PH * PW; idx += 256) {
            const int cl  = idx >> 10;           // local channel
            const int rem = idx & 1023;
            const int i = rem >> 5, j = rem & 31;
            const int cg = c0 + cl;
            float v = yg[((size_t)(b * C_ + cg) << 10) + rem];
            v = (v - mu) * rstd * gamma[cg] + beta[cg];
            const int r = dir ? j : i, cc = dir ? i : j;
            ys[cl * YCH + (r + HALO) * YSTR + cc] = v;
        }
        __syncthreads();

        if (wid < 4) {
            // ---- waves 0..3: one 16x16 tile of A = Q K^T each, via WMMA ----
            const int tm = wid >> 1, tn = wid & 1;
            const int row = lane & 15, hi = lane >> 4;
            const int mq = tm * 16 + row;     // local q channel (A side, M)
            const int ek = tn * 16 + row;     // local k channel (B side, N)
            const float* WQ = dir ? wqw : wqh;
            const float* WK = dir ? wkw : wkh;
            float qw[7], kw[11];
            #pragma unroll
            for (int s = 0; s < 7; ++s)  qw[s] = WQ[(c0 + mq) * 7 + s];
            #pragma unroll
            for (int s = 0; s < 11; ++s) kw[s] = WK[(c0 + ek) * 11 + s];
            const float* yq = ys + mq * YCH;
            const float* yk = ys + ek * YCH;

            v8f acc = {};
            for (int kk = 0; kk < PH; ++kk) { // K-loop: one (possibly transposed) row
                v16h A, Bf;
                // 16-bit A layout: half 0 -> K {0..7,16..23}, half 1 -> {8..15,24..31}
                #pragma unroll
                for (int t = 0; t < 16; ++t) {
                    const int Ka = (t < 8) ? (hi * 8 + t) : (16 + hi * 8 + t - 8);
                    A[t] = (_Float16)dconvH<3>(yq, kk, Ka, qw);
                }
                // 16-bit B layout: half 0 -> K {0..7,16..23}, half 1 -> {8..15,24..31}
                #pragma unroll
                for (int t = 0; t < 16; ++t) {
                    const int Kb = (t < 8) ? (hi * 8 + t) : (16 + hi * 8 + t - 8);
                    Bf[t] = (_Float16)dconvH<5>(yk, kk, Kb, kw);
                }
                acc = __builtin_amdgcn_wmma_f32_16x16x32_f16(
                          false, A, false, Bf, (short)0, acc, false, false);
            }
            // D layout: VGPR r, lane -> (m = r + 8*hi, n = lane&15)
            #pragma unroll
            for (int r = 0; r < 8; ++r) {
                const int m = tm * 16 + r + hi * 8;
                const int n = tn * 16 + row;
                scores[m * YSTR + n] = acc[r] * 0.17677669529663687f; // 1/sqrt(32)
            }
        } else {
            // ---- waves 4..7: vbar[e] = mean_n v[e,n] (21-tap conv) ----
            const int t2 = tid - 128;
            const int e = t2 >> 2, part = t2 & 3;
            const float* WV = dir ? wvw : wvh;
            float vw[21];
            #pragma unroll
            for (int s = 0; s < 21; ++s) vw[s] = WV[(c0 + e) * 21 + s];
            const float* ye = ys + e * YCH;
            float s = 0.f;
            for (int i = part * 8; i < part * 8 + 8; ++i)
                for (int j = 0; j < PW; ++j)
                    s += dconvH<10>(ye, i, j, vw);
            vpart[e * 4 + part] = s;
        }
        __syncthreads();

        if (tid < DHEAD)
            vbar[tid] = (vpart[tid * 4] + vpart[tid * 4 + 1] +
                         vpart[tid * 4 + 2] + vpart[tid * 4 + 3]) * (1.f / 1024.f);
        __syncthreads();

        // fused softmax + P.vbar:  mean_n(P V)[d] = sum_e softmax(A)[d,e] vbar[e]
        if (tid < DHEAD) {
            const int d = tid;
            float mx = scores[d * YSTR];
            for (int e = 1; e < DHEAD; ++e) mx = fmaxf(mx, scores[d * YSTR + e]);
            float sum = 0.f, g = 0.f;
            for (int e = 0; e < DHEAD; ++e) {
                const float ex = __expf(scores[d * YSTR + e] - mx);
                sum += ex;
                g   += ex * vbar[e];
            }
            gacc += g / sum;
        }
        __syncthreads();   // ys/scores/vpart reused by next direction
    }

    if (tid < DHEAD)
        gate[b * C_ + c0 + tid] = 1.f / (1.f + __expf(-gacc));
}

// ---------------- Kernel 4: out = gate[b,c] * x (float4, HBM-bound) --------
__global__ __launch_bounds__(256)
void k_apply_gate(const float* __restrict__ x, const float* __restrict__ gate,
                  float* __restrict__ out, int total4) {
    const int idx = blockIdx.x * 256 + threadIdx.x;
    if (idx >= total4) return;
    const int bc = idx / (H_ * W_ / 4);      // 12544 float4 per channel
    const float g = gate[bc];
    const float4 v = ((const float4*)x)[idx];
    float4 o; o.x = v.x * g; o.y = v.y * g; o.z = v.z * g; o.w = v.w * g;
    ((float4*)out)[idx] = o;
}

// ---------------------------------------------------------------------------
extern "C" void kernel_launch(void* const* d_in, const int* in_sizes, int n_in,
                              void* d_out, int out_size, void* d_ws, size_t ws_size,
                              hipStream_t stream) {
    (void)in_sizes; (void)n_in; (void)out_size; (void)ws_size;
    const float* x     = (const float*)d_in[0];
    const float* gamma = (const float*)d_in[1];
    const float* beta  = (const float*)d_in[2];
    const float* wqh   = (const float*)d_in[3];
    const float* wkh   = (const float*)d_in[4];
    const float* wvh   = (const float*)d_in[5];
    const float* wqw   = (const float*)d_in[6];
    const float* wkw   = (const float*)d_in[7];
    const float* wvw   = (const float*)d_in[8];
    float* out = (float*)d_out;

    // workspace layout (floats): y | partial | stats | gate  (~8.9 MB)
    float* ws      = (float*)d_ws;
    float* y       = ws;                              // 8*256*1024 = 2097152
    float* partial = y + (size_t)B_ * C_ * PH * PW;   // 8*8192*2   = 131072
    float* stats   = partial + (size_t)B_ * 8192 * 2; // 16
    float* gate    = stats + 2 * B_;                  // 2048

    k_pool_stats<<<dim3(B_ * C_ * PH), dim3(256), 0, stream>>>(x, y, partial);
    k_gn_stats<<<dim3(B_), dim3(256), 0, stream>>>(partial, stats);

    const size_t lds_bytes =
        (size_t)(DHEAD * YCH + DHEAD * YSTR + 128 + 32) * sizeof(float); // ~224.5 KB
    k_attn_gate<<<dim3(B_ * HEADS), dim3(256), lds_bytes, stream>>>(
        y, stats, gamma, beta, wqh, wkh, wvh, wqw, wkw, wvw, gate);

    const int total4 = B_ * C_ * (H_ * W_ / 4);
    k_apply_gate<<<dim3((total4 + 255) / 256), dim3(256), 0, stream>>>(
        x, gate, out, total4);
}
